// PcPreprocessor_9199819948254
// MI455X (gfx1250) — matrix operand order repaired and checked by reading, not compiled, not run on hardware
//
#include <hip/hip_runtime.h>
#include <hip/hip_bf16.h>
#include <stdint.h>

typedef __attribute__((ext_vector_type(16))) _Float16 v16h;
typedef __attribute__((ext_vector_type(8)))  _Float16 v8h;
typedef __attribute__((ext_vector_type(2)))  _Float16 v2h;
typedef __attribute__((ext_vector_type(8)))  float    v8f;

#define DEV __device__ __forceinline__

// ---------- order-preserving float <-> u32 (for atomic segment max) ----------
DEV unsigned fenc(float f) {
  unsigned u = __float_as_uint(f);
  return (u & 0x80000000u) ? ~u : (u | 0x80000000u);
}
DEV float fdec(unsigned u) {
  return (u & 0x80000000u) ? __uint_as_float(u & 0x7FFFFFFFu) : __uint_as_float(~u);
}

DEV int quanti(float x, float lo, float hi, float sz) {
  return (int)((x - lo) / (hi - lo) * sz);
}

// per-point 16-feature block for one scale: x = (feat@W^T+b) * (mean@Ww^T+wb)
DEV void scale_feats(float px, float py, float pz, float pw,
                     float cmx, float cmy, float cmz,
                     int xi, int yi, int zi,
                     const float* __restrict__ W,  const float* __restrict__ Bb,
                     const float* __restrict__ Ww, const float* __restrict__ Wb,
                     float* x16, float* mean6) {
  float mmx = px - cmx, mmy = py - cmy, mmz = pz - cmz;
  float gx = px - ((xi + 0.5f) * 0.2f + 0.0f);
  float gy = py - ((yi + 0.5f) * 0.2f - 25.6f);
  float gz = pz - ((zi + 0.5f) * 0.2f - 2.0f);
  float feat[10] = {px, py, pz, pw, mmx, mmy, mmz, gx, gy, gz};
  float mean[6]  = {mmx, mmy, mmz, gx, gy, gz};
  if (mean6) {
#pragma unroll
    for (int k = 0; k < 6; ++k) mean6[k] = mean[k];
  }
#pragma unroll
  for (int j = 0; j < 16; ++j) {
    float a = Bb[j];
#pragma unroll
    for (int k = 0; k < 10; ++k) a += feat[k] * W[j * 10 + k];
    float g = Wb[j];
#pragma unroll
    for (int k = 0; k < 6; ++k) g += mean[k] * Ww[j * 6 + k];
    x16[j] = a * g;
  }
}

// ---------------- pass 1: voxel ids + presence bitmaps ----------------
__global__ void k_ids(const float4* __restrict__ pc, const int* __restrict__ ind, int B,
                      int* __restrict__ lid05, int* __restrict__ lid1,
                      unsigned* __restrict__ bm05, unsigned* __restrict__ bm1, int n) {
  int i = blockIdx.x * blockDim.x + threadIdx.x;
  if (i >= n) return;
  __builtin_prefetch(&pc[i + 2048], 0, 1);   // global_prefetch_b8, HBM stream
  float4 p = pc[i];
  int bt = 0;
  for (int b = 1; b < B; ++b) if (i >= ind[b]) bt = b;
  int x5 = quanti(p.x, 0.0f, 51.2f, 512.f);
  int y5 = quanti(p.y, -25.6f, 25.6f, 512.f);
  int z5 = quanti(p.z, -2.0f, 4.4f, 64.f);
  int l5 = ((bt * 512 + x5) * 512 + y5) * 64 + z5;
  int x1 = quanti(p.x, 0.0f, 51.2f, 256.f);
  int y1 = quanti(p.y, -25.6f, 25.6f, 256.f);
  int z1 = quanti(p.z, -2.0f, 4.4f, 32.f);
  int l1 = ((bt * 256 + x1) * 256 + y1) * 32 + z1;
  lid05[i] = l5;
  lid1[i]  = l1;
  atomicOr(&bm05[l5 >> 5], 1u << (l5 & 31));
  atomicOr(&bm1[l1 >> 5], 1u << (l1 & 31));
}

// ---------------- popcount prefix scan over bitmap words ----------------
__global__ void k_blockpop(const unsigned* __restrict__ bm, unsigned* __restrict__ bsum, int W) {
  __shared__ unsigned red[256];
  int base = blockIdx.x * 1024;
  unsigned s = 0;
  for (int j = threadIdx.x; j < 1024; j += 256) {
    int w = base + j;
    if (w < W) s += (unsigned)__popc(bm[w]);
  }
  red[threadIdx.x] = s;
  __syncthreads();
  for (int off = 128; off > 0; off >>= 1) {
    if ((int)threadIdx.x < off) red[threadIdx.x] += red[threadIdx.x + off];
    __syncthreads();
  }
  if (threadIdx.x == 0) bsum[blockIdx.x] = red[0];
}

__global__ void k_scan_bsum(unsigned* __restrict__ bsum, int nb) {
  __shared__ unsigned sh[1024];
  int t = threadIdx.x;
  unsigned v = (t < nb) ? bsum[t] : 0u;
  sh[t] = v;
  __syncthreads();
  for (int off = 1; off < 1024; off <<= 1) {
    unsigned add = (t >= off) ? sh[t - off] : 0u;
    __syncthreads();
    sh[t] += add;
    __syncthreads();
  }
  if (t < nb) bsum[t] = sh[t] - v;  // exclusive
}

__global__ void k_wpfx(const unsigned* __restrict__ bm, const unsigned* __restrict__ bsum,
                       unsigned* __restrict__ wpfx, int W) {
  __shared__ unsigned sh[1024];
  int t = threadIdx.x;
  int w = blockIdx.x * 1024 + t;
  unsigned p = (w < W) ? (unsigned)__popc(bm[w]) : 0u;
  sh[t] = p;
  __syncthreads();
  for (int off = 1; off < 1024; off <<= 1) {
    unsigned add = (t >= off) ? sh[t - off] : 0u;
    __syncthreads();
    sh[t] += add;
    __syncthreads();
  }
  if (w < W) wpfx[w] = sh[t] - p + bsum[blockIdx.x];
}

// ---------------- pass 2: inverse ids + segment stats + coord scatter ----------------
__global__ void k_inv(const float4* __restrict__ pc, const int* __restrict__ ind, int B,
                      const int* __restrict__ lid05, const int* __restrict__ lid1,
                      const unsigned* __restrict__ bm05, const unsigned* __restrict__ wp05,
                      const unsigned* __restrict__ bm1, const unsigned* __restrict__ wp1,
                      int* __restrict__ inv05, int* __restrict__ inv1,
                      float* __restrict__ cnt05, float* __restrict__ sum05,
                      float* __restrict__ cnt1, float* __restrict__ sum1,
                      int* __restrict__ coordOut, int* __restrict__ invOut, int n) {
  int i = blockIdx.x * blockDim.x + threadIdx.x;
  if (i >= n) return;
  int l5 = lid05[i];
  unsigned w5 = (unsigned)l5 >> 5, b5 = (unsigned)l5 & 31u;
  int v5 = (int)(wp05[w5] + (unsigned)__popc(bm05[w5] & ((1u << b5) - 1u)));
  int l1 = lid1[i];
  unsigned w1 = (unsigned)l1 >> 5, b1 = (unsigned)l1 & 31u;
  int v1 = (int)(wp1[w1] + (unsigned)__popc(bm1[w1] & ((1u << b1) - 1u)));
  inv05[i] = v5;
  inv1[i]  = v1;
  invOut[i] = v1;
  float4 p = pc[i];
  atomicAdd(&cnt05[v5], 1.0f);
  atomicAdd(&sum05[3 * v5 + 0], p.x);
  atomicAdd(&sum05[3 * v5 + 1], p.y);
  atomicAdd(&sum05[3 * v5 + 2], p.z);
  atomicAdd(&cnt1[v1], 1.0f);
  atomicAdd(&sum1[3 * v1 + 0], p.x);
  atomicAdd(&sum1[3 * v1 + 1], p.y);
  atomicAdd(&sum1[3 * v1 + 2], p.z);
  int bt = 0;
  for (int b = 1; b < B; ++b) if (i >= ind[b]) bt = b;
  int x1 = quanti(p.x, 0.0f, 51.2f, 256.f);
  int y1 = quanti(p.y, -25.6f, 25.6f, 256.f);
  int z1 = quanti(p.z, -2.0f, 4.4f, 32.f);
  coordOut[4 * v1 + 0] = bt;   // duplicates write identical values
  coordOut[4 * v1 + 1] = x1;
  coordOut[4 * v1 + 2] = y1;
  coordOut[4 * v1 + 3] = z1;
}

// ---------------- pass 3: per-voxel max of the 16-d scale features ----------------
__global__ void k_mx(const float4* __restrict__ pc,
                     const int* __restrict__ inv05, const int* __restrict__ inv1,
                     const float* __restrict__ cnt05, const float* __restrict__ sum05,
                     const float* __restrict__ cnt1, const float* __restrict__ sum1,
                     const float* __restrict__ w05, const float* __restrict__ b05,
                     const float* __restrict__ ww05, const float* __restrict__ wb05,
                     const float* __restrict__ w1, const float* __restrict__ b1,
                     const float* __restrict__ ww1, const float* __restrict__ wb1,
                     unsigned* __restrict__ mx05, unsigned* __restrict__ mx1, int n) {
  int i = blockIdx.x * blockDim.x + threadIdx.x;
  if (i >= n) return;
  float4 p = pc[i];
  int s5 = inv05[i], s1 = inv1[i];
  float c5 = cnt05[s5], c1 = cnt1[s1];
  float x16[16];
  scale_feats(p.x, p.y, p.z, p.w,
              sum05[3 * s5 + 0] / c5, sum05[3 * s5 + 1] / c5, sum05[3 * s5 + 2] / c5,
              quanti(p.x, 0.0f, 51.2f, 512.f), quanti(p.y, -25.6f, 25.6f, 512.f),
              quanti(p.z, -2.0f, 4.4f, 64.f), w05, b05, ww05, wb05, x16, nullptr);
#pragma unroll
  for (int j = 0; j < 16; ++j) atomicMax(&mx05[(size_t)s5 * 16 + j], fenc(x16[j]));
  scale_feats(p.x, p.y, p.z, p.w,
              sum1[3 * s1 + 0] / c1, sum1[3 * s1 + 1] / c1, sum1[3 * s1 + 2] / c1,
              quanti(p.x, 0.0f, 51.2f, 256.f), quanti(p.y, -25.6f, 25.6f, 256.f),
              quanti(p.z, -2.0f, 4.4f, 32.f), w1, b1, ww1, wb1, x16, nullptr);
#pragma unroll
  for (int j = 0; j < 16; ++j) atomicMax(&mx1[(size_t)s1 * 16 + j], fenc(x16[j]));
}

// ---------------- pass 4 (WMMA tail): agg MLP + segment-max scatter ----------------
// 256 threads = 8 wave32 tiles per block. Weights staged once per block in
// per-lane WMMA fragment order so every B fragment is a single aligned 32-byte
// LDS read (ds_load_b128 pair) and every A fragment is two b128 reads.
__global__ void __launch_bounds__(256)
k_tail(const float4* __restrict__ pc,
       const int* __restrict__ inv05, const int* __restrict__ inv1,
       const float* __restrict__ cnt05, const float* __restrict__ sum05,
       const float* __restrict__ cnt1, const float* __restrict__ sum1,
       const float* __restrict__ w05, const float* __restrict__ b05,
       const float* __restrict__ ww05, const float* __restrict__ wb05,
       const float* __restrict__ w1, const float* __restrict__ b1,
       const float* __restrict__ ww1, const float* __restrict__ wb1,
       const float* __restrict__ am_w, const float* __restrict__ am_b,
       const float* __restrict__ at_w, const float* __restrict__ at_b,
       const float* __restrict__ af_w, const float* __restrict__ af_b,
       const unsigned* __restrict__ mx05, const unsigned* __restrict__ mx1,
       unsigned* __restrict__ maxseg, int n) {
  // weight fragments, shared by all 8 waves
  __shared__ __align__(32) _Float16 sBtF[2][2][32][16];  // [kchunk][ntile][lane][elem]
  __shared__ __align__(32) _Float16 sBfF[2][32][16];     // [ntile][lane][elem]
  __shared__ float sAtB[32];
  __shared__ float sAfB[32];
  // per-wave staging
  __shared__ __align__(16) _Float16 sCat[8][16][64];
  __shared__ __align__(16) _Float16 sHh[8][16][32];
  __shared__ float sA[8][16][32];
  __shared__ int   sInv[8][16];

  const int tid = threadIdx.x;
  const int lane = tid & 31;
  const int wv = tid >> 5;

  // stage at_w fragments: element e of lane l, chunk kc, ntile nt  <->  K=(l>>4)*8+(e&7)+(e>>3)*16+kc*32, N=(l&15)+nt*16
  for (int idx = tid; idx < 2 * 2 * 32 * 16; idx += 256) {
    int e = idx & 15, l = (idx >> 4) & 31, nt = (idx >> 9) & 1, kc = (idx >> 10) & 1;
    int k = ((l >> 4) * 8) + (e & 7) + ((e >> 3) * 16) + kc * 32;
    int nn2 = (l & 15) + nt * 16;
    sBtF[kc][nt][l][e] = (_Float16)at_w[nn2 * 64 + k];
  }
  for (int idx = tid; idx < 2 * 32 * 16; idx += 256) {
    int e = idx & 15, l = (idx >> 4) & 31, nt = (idx >> 9) & 1;
    int k = ((l >> 4) * 8) + (e & 7) + ((e >> 3) * 16);
    int nn2 = (l & 15) + nt * 16;
    sBfF[nt][l][e] = (_Float16)af_w[nn2 * 32 + k];
  }
  if (tid < 32) {
    sAtB[tid] = at_b[tid];
    sAfB[tid] = af_b[tid];
  }
  __syncthreads();

  const int tile = blockIdx.x * 8 + wv;
  if (lane < 16) {
    int i = tile * 16 + lane;
    if (i >= n) i = n - 1;              // clamp: duplicate max is idempotent
    float4 p = pc[i];
    int s5 = inv05[i], s1 = inv1[i];
    float c5 = cnt05[s5], c1 = cnt1[s1];
    float x16[16], mean1[6];
    scale_feats(p.x, p.y, p.z, p.w,
                sum05[3 * s5 + 0] / c5, sum05[3 * s5 + 1] / c5, sum05[3 * s5 + 2] / c5,
                quanti(p.x, 0.0f, 51.2f, 512.f), quanti(p.y, -25.6f, 25.6f, 512.f),
                quanti(p.z, -2.0f, 4.4f, 64.f), w05, b05, ww05, wb05, x16, nullptr);
#pragma unroll
    for (int j = 0; j < 16; j += 2) {
      v2h t = {(_Float16)x16[j], (_Float16)x16[j + 1]};
      *(v2h*)&sCat[wv][lane][j] = t;
      v2h u = {(_Float16)fdec(mx05[(size_t)s5 * 16 + j]),
               (_Float16)fdec(mx05[(size_t)s5 * 16 + j + 1])};
      *(v2h*)&sCat[wv][lane][16 + j] = u;
    }
    scale_feats(p.x, p.y, p.z, p.w,
                sum1[3 * s1 + 0] / c1, sum1[3 * s1 + 1] / c1, sum1[3 * s1 + 2] / c1,
                quanti(p.x, 0.0f, 51.2f, 256.f), quanti(p.y, -25.6f, 25.6f, 256.f),
                quanti(p.z, -2.0f, 4.4f, 32.f), w1, b1, ww1, wb1, x16, mean1);
#pragma unroll
    for (int j = 0; j < 16; j += 2) {
      v2h t = {(_Float16)x16[j], (_Float16)x16[j + 1]};
      *(v2h*)&sCat[wv][lane][32 + j] = t;
      v2h u = {(_Float16)fdec(mx1[(size_t)s1 * 16 + j]),
               (_Float16)fdec(mx1[(size_t)s1 * 16 + j + 1])};
      *(v2h*)&sCat[wv][lane][48 + j] = u;
    }
#pragma unroll
    for (int j = 0; j < 32; ++j) {
      float g = am_b[j];
#pragma unroll
      for (int k = 0; k < 6; ++k) g += mean1[k] * am_w[j * 6 + k];
      sA[wv][lane][j] = fmaxf(g, 0.0f);
    }
    sInv[wv][lane] = s1;
  }
  // same-wave LDS producer->consumer: DS ops are in-order within a wave;
  // the compiler inserts the needed s_wait_dscnt on the loads below.

  // A fragments: 16x32 f16, row m=lane&15; each is two contiguous 8-half runs
  const int m  = lane & 15;
  const int kb = (lane >> 4) * 8;
  v8h a0lo = *(const v8h*)&sCat[wv][m][kb];
  v8h a0hi = *(const v8h*)&sCat[wv][m][16 + kb];
  v8h a1lo = *(const v8h*)&sCat[wv][m][32 + kb];
  v8h a1hi = *(const v8h*)&sCat[wv][m][48 + kb];
  v16h A0 = __builtin_shufflevector(a0lo, a0hi, 0, 1, 2, 3, 4, 5, 6, 7, 8, 9, 10, 11, 12, 13, 14, 15);
  v16h A1 = __builtin_shufflevector(a1lo, a1hi, 0, 1, 2, 3, 4, 5, 6, 7, 8, 9, 10, 11, 12, 13, 14, 15);
  // B fragments: single aligned 32B reads (pre-staged in fragment order)
  v16h B00 = *(const v16h*)&sBtF[0][0][lane][0];
  v16h B01 = *(const v16h*)&sBtF[0][1][lane][0];
  v16h B10 = *(const v16h*)&sBtF[1][0][lane][0];
  v16h B11 = *(const v16h*)&sBtF[1][1][lane][0];

  v8f acc0 = {}, acc1 = {};
  acc0 = __builtin_amdgcn_wmma_f32_16x16x32_f16(false, A0, false, B00, (short)0, acc0, false, false);
  acc0 = __builtin_amdgcn_wmma_f32_16x16x32_f16(false, A1, false, B10, (short)0, acc0, false, false);
  acc1 = __builtin_amdgcn_wmma_f32_16x16x32_f16(false, A0, false, B01, (short)0, acc1, false, false);
  acc1 = __builtin_amdgcn_wmma_f32_16x16x32_f16(false, A1, false, B11, (short)0, acc1, false, false);

  // epilogue 1: h = relu(t + at_b) * gate  (C/D layout: row = r + 8*(lane>>4), col = lane&15)
  const int nn = lane & 15;
#pragma unroll
  for (int r = 0; r < 8; ++r) {
    int mrow = r + 8 * (lane >> 4);
    float t0 = fmaxf(acc0[r] + sAtB[nn], 0.0f);
    sHh[wv][mrow][nn] = (_Float16)(t0 * sA[wv][mrow][nn]);
    float t1 = fmaxf(acc1[r] + sAtB[nn + 16], 0.0f);
    sHh[wv][mrow][nn + 16] = (_Float16)(t1 * sA[wv][mrow][nn + 16]);
  }

  // second GEMM: h[16x32] @ af_w^T (K=32, one chunk, two N tiles)
  v8h a2lo = *(const v8h*)&sHh[wv][m][kb];
  v8h a2hi = *(const v8h*)&sHh[wv][m][16 + kb];
  v16h A2 = __builtin_shufflevector(a2lo, a2hi, 0, 1, 2, 3, 4, 5, 6, 7, 8, 9, 10, 11, 12, 13, 14, 15);
  v16h Bf0 = *(const v16h*)&sBfF[0][lane][0];
  v16h Bf1 = *(const v16h*)&sBfF[1][lane][0];

  v8f d0 = {}, d1 = {};
  d0 = __builtin_amdgcn_wmma_f32_16x16x32_f16(false, A2, false, Bf0, (short)0, d0, false, false);
  d1 = __builtin_amdgcn_wmma_f32_16x16x32_f16(false, A2, false, Bf1, (short)0, d1, false, false);

#pragma unroll
  for (int r = 0; r < 8; ++r) {
    int mrow = r + 8 * (lane >> 4);
    int seg = sInv[wv][mrow];
    atomicMax(&maxseg[(size_t)seg * 32 + nn],      fenc(d0[r] + sAfB[nn]));
    atomicMax(&maxseg[(size_t)seg * 32 + nn + 16], fenc(d1[r] + sAfB[nn + 16]));
  }
}

// ---------------- pass 5: decode maxseg into output (0 for empty rows) ----------------
__global__ void k_final(const unsigned* __restrict__ maxseg, const float* __restrict__ cnt1,
                        float* __restrict__ outMax, int n) {
  int idx = blockIdx.x * blockDim.x + threadIdx.x;
  if (idx >= n * 32) return;
  int row = idx >> 5;
  outMax[idx] = (cnt1[row] > 0.0f) ? fdec(maxseg[idx]) : 0.0f;
}

extern "C" void kernel_launch(void* const* d_in, const int* in_sizes, int n_in,
                              void* d_out, int out_size, void* d_ws, size_t ws_size,
                              hipStream_t stream) {
  const float4* pc  = (const float4*)d_in[0];
  const int*    ind = (const int*)d_in[1];
  const float *w05 = (const float*)d_in[2],  *b05 = (const float*)d_in[3];
  const float *ww05 = (const float*)d_in[4], *wb05 = (const float*)d_in[5];
  const float *w1 = (const float*)d_in[6],   *b1 = (const float*)d_in[7];
  const float *ww1 = (const float*)d_in[8],  *wb1 = (const float*)d_in[9];
  const float *am_w = (const float*)d_in[10], *am_b = (const float*)d_in[11];
  const float *at_w = (const float*)d_in[12], *at_b = (const float*)d_in[13];
  const float *af_w = (const float*)d_in[14], *af_b = (const float*)d_in[15];

  const size_t N = (size_t)in_sizes[0] / 4;
  const int B = in_sizes[1] - 1;
  const size_t R05 = (size_t)B * 512 * 512 * 64;
  const size_t R1  = (size_t)B * 256 * 256 * 32;
  const size_t W05w = (R05 + 31) / 32;
  const size_t W1w  = (R1 + 31) / 32;
  const int nb05 = (int)((W05w + 1023) / 1024);
  const int nb1  = (int)((W1w + 1023) / 1024);

  char* p = (char*)d_ws;
  auto take = [&](size_t bytes) -> void* {
    void* r = (void*)p;
    p += (bytes + 255) & ~(size_t)255;
    return r;
  };
  int* lid05 = (int*)take(N * 4);
  int* lid1  = (int*)take(N * 4);
  int* inv05 = (int*)take(N * 4);
  int* inv1  = (int*)take(N * 4);
  unsigned* bm05 = (unsigned*)take(W05w * 4);
  unsigned* bm1  = (unsigned*)take(W1w * 4);
  unsigned* wp05 = (unsigned*)take(W05w * 4);
  unsigned* wp1  = (unsigned*)take(W1w * 4);
  unsigned* bs05 = (unsigned*)take((size_t)nb05 * 4);
  unsigned* bs1  = (unsigned*)take((size_t)nb1 * 4);
  float* cnt05 = (float*)take(N * 4);
  float* sum05 = (float*)take(N * 12);
  float* cnt1  = (float*)take(N * 4);
  float* sum1  = (float*)take(N * 12);
  unsigned* mx05 = (unsigned*)take(N * 64);
  unsigned* mx1  = (unsigned*)take(N * 64);
  unsigned* mseg = (unsigned*)take(N * 128);

  float* outMax = (float*)d_out;
  int* outCoord = (int*)(outMax + N * 32);
  int* outInv   = (int*)(outMax + N * 36);

  // zero init (fenc of any finite value > 0, so 0 doubles as -inf for the max arrays)
  hipMemsetAsync(bm05, 0, W05w * 4, stream);
  hipMemsetAsync(bm1, 0, W1w * 4, stream);
  hipMemsetAsync(cnt05, 0, N * 4, stream);
  hipMemsetAsync(sum05, 0, N * 12, stream);
  hipMemsetAsync(cnt1, 0, N * 4, stream);
  hipMemsetAsync(sum1, 0, N * 12, stream);
  hipMemsetAsync(mx05, 0, N * 64, stream);
  hipMemsetAsync(mx1, 0, N * 64, stream);
  hipMemsetAsync(mseg, 0, N * 128, stream);
  hipMemsetAsync(outCoord, 0, N * 16, stream);

  const int T = 256;
  const int G = (int)((N + T - 1) / T);
  k_ids<<<G, T, 0, stream>>>(pc, ind, B, lid05, lid1, bm05, bm1, (int)N);
  k_blockpop<<<nb05, 256, 0, stream>>>(bm05, bs05, (int)W05w);
  k_blockpop<<<nb1, 256, 0, stream>>>(bm1, bs1, (int)W1w);
  k_scan_bsum<<<1, 1024, 0, stream>>>(bs05, nb05);
  k_scan_bsum<<<1, 1024, 0, stream>>>(bs1, nb1);
  k_wpfx<<<nb05, 1024, 0, stream>>>(bm05, bs05, wp05, (int)W05w);
  k_wpfx<<<nb1, 1024, 0, stream>>>(bm1, bs1, wp1, (int)W1w);
  k_inv<<<G, T, 0, stream>>>(pc, ind, B, lid05, lid1, bm05, wp05, bm1, wp1,
                             inv05, inv1, cnt05, sum05, cnt1, sum1, outCoord, outInv, (int)N);
  k_mx<<<G, T, 0, stream>>>(pc, inv05, inv1, cnt05, sum05, cnt1, sum1,
                            w05, b05, ww05, wb05, w1, b1, ww1, wb1, mx05, mx1, (int)N);
  k_tail<<<(int)((N + 127) / 128), 256, 0, stream>>>(pc, inv05, inv1, cnt05, sum05, cnt1, sum1,
                                                     w05, b05, ww05, wb05, w1, b1, ww1, wb1,
                                                     am_w, am_b, at_w, at_b, af_w, af_b,
                                                     mx05, mx1, mseg, (int)N);
  k_final<<<(int)((N * 32 + 255) / 256), 256, 0, stream>>>(mseg, cnt1, outMax, (int)N);
}
